// GRPOMixtureOfExperts_70403103916702
// MI455X (gfx1250) — compile-verified, matrix operand order checked by reading, not run on hardware
//
#include <hip/hip_runtime.h>
#include <math.h>

#define Tn 8192   // B*S tokens
#define Dd 1024
#define Hh 4096
#define Ee 8
#define Kk 2

typedef __attribute__((ext_vector_type(16))) __bf16 v16bf;
typedef __attribute__((ext_vector_type(8)))  float  v8f;

__device__ __forceinline__ unsigned short f2bf(float f) {
    unsigned u = __float_as_uint(f);
    unsigned r = u + 0x7FFFu + ((u >> 16) & 1u);   // round-to-nearest-even
    return (unsigned short)(r >> 16);
}

__device__ __forceinline__ float gelu_exact(float v) {
    return 0.5f * v * (1.0f + erff(v * 0.70710678118654752f));
}

// Build a v16bf WMMA fragment from an LDS row holding 32 contiguous-K bf16.
// ISA layout (16-bit A/B 16x32): lanes 0-15 hold K {0..7,16..23},
// lanes 16-31 hold K {8..15,24..31}. Two 16B chunks -> two ds_load_b128.
__device__ __forceinline__ v16bf load_frag(const unsigned short* rowp, int lane_hi) {
    const uint4* q = (const uint4*)(rowp + (lane_hi ? 8 : 0));
    union U { uint4 q[2]; v16bf v; } u;
    u.q[0] = q[0];   // K chunk at +0
    u.q[1] = q[2];   // K chunk at +16 elements
    return u.v;
}

__device__ __forceinline__ v8f wmma_bf16(v16bf a, v16bf b, v8f c) {
    return __builtin_amdgcn_wmma_f32_16x16x32_bf16(
        false, a, false, b, (short)0, c, false, false);
}

// ---------------------------------------------------------------- init
__global__ void init_cnt_kernel(int* cnt) {
    if (threadIdx.x < Ee) cnt[threadIdx.x] = 0;
}

// ---------------------------------------------------------------- gating
__global__ void gate_kernel(const float* __restrict__ x,
                            const float* __restrict__ gw,
                            const float* __restrict__ gb,
                            float* __restrict__ topk_w,
                            int* __restrict__ list,
                            int* __restrict__ cnt) {
    int wave = threadIdx.x >> 5;
    int lane = threadIdx.x & 31;
    int t = blockIdx.x * 8 + wave;
    if (t >= Tn) return;

    float acc[Ee];
#pragma unroll
    for (int e = 0; e < Ee; ++e) acc[e] = 0.0f;

    const float* xr = x + (size_t)t * Dd;
    for (int d = lane; d < Dd; d += 32) {
        float xv = xr[d];
        const float* g = gw + (size_t)d * Ee;
#pragma unroll
        for (int e = 0; e < Ee; ++e) acc[e] += xv * g[e];
    }
#pragma unroll
    for (int e = 0; e < Ee; ++e) {
#pragma unroll
        for (int off = 16; off > 0; off >>= 1)
            acc[e] += __shfl_xor(acc[e], off, 32);
    }
    if (lane == 0) {
        float sc[Ee];
#pragma unroll
        for (int e = 0; e < Ee; ++e) sc[e] = acc[e] + gb[e];
        int i0 = 0;
#pragma unroll
        for (int e = 1; e < Ee; ++e) if (sc[e] > sc[i0]) i0 = e;
        int i1 = (i0 == 0) ? 1 : 0;
#pragma unroll
        for (int e = 0; e < Ee; ++e)
            if (e != i0 && sc[e] > sc[i1]) i1 = e;
        float e1 = __expf(sc[i1] - sc[i0]);
        float inv = 1.0f / (1.0f + e1);
        topk_w[t * 2 + 0] = inv;
        topk_w[t * 2 + 1] = e1 * inv;
        int p0 = atomicAdd(&cnt[i0], 1);
        list[i0 * Tn + p0] = t * 2 + 0;
        int p1 = atomicAdd(&cnt[i1], 1);
        list[i1 * Tn + p1] = t * 2 + 1;
    }
}

// ---------------------------------------------------------------- f32 -> bf16
__global__ void cvt_bf16_kernel(const float* __restrict__ src,
                                unsigned short* __restrict__ dst, int n2) {
    int i = blockIdx.x * blockDim.x + threadIdx.x;
    if (i < n2) {
        float2 f = *(const float2*)(src + (size_t)i * 2);
        unsigned v = (unsigned)f2bf(f.x) | ((unsigned)f2bf(f.y) << 16);
        *(unsigned*)(dst + (size_t)i * 2) = v;
    }
}

// ---------------------------------------------------------------- transpose + cvt
// src: [E][R][C] f32  ->  dst: [E][C][R] bf16   (32x32 LDS tiles)
__global__ void transpose_cvt_kernel(const float* __restrict__ src,
                                     unsigned short* __restrict__ dst,
                                     int R, int C) {
    __shared__ unsigned short tile[32][33];
    int e = blockIdx.z;
    src += (size_t)e * R * C;
    dst += (size_t)e * R * C;
    int c0 = blockIdx.x * 32, r0 = blockIdx.y * 32;
    int tx = threadIdx.x, ty = threadIdx.y;      // block (32,8)
#pragma unroll
    for (int i = 0; i < 32; i += 8)
        tile[ty + i][tx] = f2bf(src[(size_t)(r0 + ty + i) * C + c0 + tx]);
    __syncthreads();
#pragma unroll
    for (int i = 0; i < 32; i += 8)
        dst[(size_t)(c0 + ty + i) * R + r0 + tx] = tile[tx][ty + i];
}

// ---------------------------------------------------------------- grouped WMMA GEMM
// Block tile 128(M) x 256(N), K-step 32, double-buffered LDS (1 barrier/step).
// Wave tile 32x128: 2 A-frags x 8 B-frags -> 16 v_wmma per K-step.
// STAGE 1: h = gelu(x_bf16[token] @ w1t^T + b1)  -> hbuf bf16   (A row = slot>>1)
// STAGE 2: y = hbuf[slot] @ w2t^T + b2          -> ybuf f32    (A row = slot)
// Bt layout: [E][Ntot][KD] (pre-transposed so B frags are contiguous along K).
template <int KD, int STAGE>
__global__ __launch_bounds__(256)
void moe_gemm_kernel(const unsigned short* __restrict__ Abase,
                     const unsigned short* __restrict__ Bt,
                     const float* __restrict__ bias,
                     const int* __restrict__ list,
                     const int* __restrict__ cnt,
                     unsigned short* __restrict__ out_bf,
                     float* __restrict__ out_f,
                     int Ntot) {
    constexpr int LDSTR = 40;      // 80B LDS row stride: 16B-aligned, bank-conflict-free
    constexpr int ASZ = 128 * LDSTR;
    constexpr int BSZ = 256 * LDSTR;
    constexpr int NSTEP = KD / 32;
    __shared__ unsigned short As[2 * ASZ];
    __shared__ unsigned short Bs[2 * BSZ];
    __shared__ int slots[128];

    int e  = blockIdx.z;
    int m0 = blockIdx.y * 128;
    int n0 = blockIdx.x * 256;
    int nrows_e = cnt[e];
    if (m0 >= nrows_e) return;                  // routed early-exit (block-uniform)
    int nrow = nrows_e - m0; if (nrow > 128) nrow = 128;

    int tid = threadIdx.x;
    if (tid < 128)
        slots[tid] = (tid < nrow) ? list[e * Tn + m0 + tid] : 0;
    __syncthreads();

    // ---- staging assignments
    // A: 2 threads per row, 16 bf16 each.  B: 1 thread per row, 32 bf16.
    int lr = tid >> 1;
    int lh = (tid & 1) * 16;
    int sA = slots[lr];
    const unsigned short* arow =
        (STAGE == 1) ? (Abase + (size_t)(sA >> 1) * KD + lh)
                     : (Abase + (size_t)sA * KD + lh);
    const unsigned short* brow = Bt + ((size_t)e * Ntot + n0 + tid) * KD;
    int aoff = lr * LDSTR + lh;
    int boff = tid * LDSTR;

    // ---- per-wave tile
    int wave = tid >> 5, lane = tid & 31;
    int wm = wave & 3;                          // row group (x32)
    int wn = wave >> 2;                         // col group (x128)
    int lane_hi = (lane >= 16);
    int l15 = lane & 15;

    v8f zacc = {};
    v8f acc[2][8];
#pragma unroll
    for (int i = 0; i < 2; ++i)
#pragma unroll
        for (int j = 0; j < 8; ++j) acc[i][j] = zacc;

    // ---- preload K-step 0 into buffer 0
    {
        uint4 pa0 = *(const uint4*)(arow);
        uint4 pa1 = *(const uint4*)(arow + 8);
        uint4 pb0 = *(const uint4*)(brow);
        uint4 pb1 = *(const uint4*)(brow + 8);
        uint4 pb2 = *(const uint4*)(brow + 16);
        uint4 pb3 = *(const uint4*)(brow + 24);
        *(uint4*)(As + aoff)     = pa0;
        *(uint4*)(As + aoff + 8) = pa1;
        *(uint4*)(Bs + boff)      = pb0;
        *(uint4*)(Bs + boff + 8)  = pb1;
        *(uint4*)(Bs + boff + 16) = pb2;
        *(uint4*)(Bs + boff + 24) = pb3;
    }
    __syncthreads();

    for (int it = 0; it < NSTEP; ++it) {
        int cur = it & 1;
        bool has_next = (it + 1) < NSTEP;

        // prefetch next K-step from global into registers (latency hidden by WMMAs)
        uint4 pa0, pa1, pb0, pb1, pb2, pb3;
        if (has_next) {
            int kk = (it + 1) * 32;
            pa0 = *(const uint4*)(arow + kk);
            pa1 = *(const uint4*)(arow + kk + 8);
            pb0 = *(const uint4*)(brow + kk);
            pb1 = *(const uint4*)(brow + kk + 8);
            pb2 = *(const uint4*)(brow + kk + 16);
            pb3 = *(const uint4*)(brow + kk + 24);
        }

        // compute from current buffer
        const unsigned short* Asb = As + cur * ASZ;
        const unsigned short* Bsb = Bs + cur * BSZ;
        v16bf afrag0 = load_frag(&Asb[(wm * 32 + l15) * LDSTR], lane_hi);
        v16bf afrag1 = load_frag(&Asb[(wm * 32 + 16 + l15) * LDSTR], lane_hi);
#pragma unroll
        for (int j = 0; j < 8; ++j) {
            v16bf bfrag = load_frag(&Bsb[(wn * 128 + j * 16 + l15) * LDSTR], lane_hi);
            acc[0][j] = wmma_bf16(afrag0, bfrag, acc[0][j]);
            acc[1][j] = wmma_bf16(afrag1, bfrag, acc[1][j]);
        }

        // store prefetched data into the other buffer
        if (has_next) {
            unsigned short* Asn = As + (cur ^ 1) * ASZ;
            unsigned short* Bsn = Bs + (cur ^ 1) * BSZ;
            *(uint4*)(Asn + aoff)     = pa0;
            *(uint4*)(Asn + aoff + 8) = pa1;
            *(uint4*)(Bsn + boff)      = pb0;
            *(uint4*)(Bsn + boff + 8)  = pb1;
            *(uint4*)(Bsn + boff + 16) = pb2;
            *(uint4*)(Bsn + boff + 24) = pb3;
        }
        __syncthreads();   // single barrier per K-step (double buffer)
    }

    // ---- epilogue (C layout: lanes 0-15 -> M=v, lanes 16-31 -> M=v+8)
#pragma unroll
    for (int i = 0; i < 2; ++i) {
#pragma unroll
        for (int j = 0; j < 8; ++j) {
            int col = n0 + wn * 128 + j * 16 + l15;
            float bv = bias[(size_t)e * Ntot + col];
#pragma unroll
            for (int v = 0; v < 8; ++v) {
                int r = wm * 32 + i * 16 + v + (lane_hi ? 8 : 0);
                if (r < nrow) {
                    int s = slots[r];
                    float val = acc[i][j][v] + bv;
                    if (STAGE == 1) {
                        out_bf[(size_t)s * Ntot + col] = f2bf(gelu_exact(val));
                    } else {
                        out_f[(size_t)s * Ntot + col] = val;
                    }
                }
            }
        }
    }
}

// ---------------------------------------------------------------- combine
// out[t] = g0 * y[2t] + g1 * y[2t+1]
__global__ void combine_kernel(const float* __restrict__ ybuf,
                               const float* __restrict__ topk_w,
                               float* __restrict__ out, int total) {
    int i = (blockIdx.x * blockDim.x + threadIdx.x) * 4;
    if (i >= total) return;
    int t = i / Dd;
    int c = i - t * Dd;
    float w0 = topk_w[t * 2 + 0];
    float w1 = topk_w[t * 2 + 1];
    float4 y0 = *(const float4*)(ybuf + (size_t)(t * 2 + 0) * Dd + c);
    float4 y1 = *(const float4*)(ybuf + (size_t)(t * 2 + 1) * Dd + c);
    float4 o;
    o.x = w0 * y0.x + w1 * y1.x;
    o.y = w0 * y0.y + w1 * y1.y;
    o.z = w0 * y0.z + w1 * y1.z;
    o.w = w0 * y0.w + w1 * y1.w;
    *(float4*)(out + i) = o;
}

// ================================================================ launch
extern "C" void kernel_launch(void* const* d_in, const int* in_sizes, int n_in,
                              void* d_out, int out_size, void* d_ws, size_t ws_size,
                              hipStream_t stream) {
    (void)in_sizes; (void)n_in; (void)out_size; (void)ws_size;
    const float* x  = (const float*)d_in[0];
    const float* gw = (const float*)d_in[1];
    const float* gb = (const float*)d_in[2];
    const float* w1 = (const float*)d_in[3];
    const float* b1 = (const float*)d_in[4];
    const float* w2 = (const float*)d_in[5];
    const float* b2 = (const float*)d_in[6];
    float* out = (float*)d_out;

    char* ws = (char*)d_ws;
    size_t off = 0;
    auto wsalloc = [&](size_t bytes) -> void* {
        void* p = ws + off;
        off = (off + bytes + 255) & ~((size_t)255);
        return p;
    };
    unsigned short* xbf  = (unsigned short*)wsalloc((size_t)Tn * Dd * 2);
    unsigned short* w1t  = (unsigned short*)wsalloc((size_t)Ee * Hh * Dd * 2);
    unsigned short* w2t  = (unsigned short*)wsalloc((size_t)Ee * Dd * Hh * 2);
    unsigned short* hbuf = (unsigned short*)wsalloc((size_t)Tn * Kk * Hh * 2);
    float*          ybuf = (float*)wsalloc((size_t)Tn * Kk * Dd * 4);
    float*        topk_w = (float*)wsalloc((size_t)Tn * Kk * 4);
    int*            list = (int*)wsalloc((size_t)Ee * Tn * 4);
    int*             cnt = (int*)wsalloc((size_t)Ee * 4);

    init_cnt_kernel<<<1, 32, 0, stream>>>(cnt);
    gate_kernel<<<Tn / 8, 256, 0, stream>>>(x, gw, gb, topk_w, list, cnt);

    // conversions (independent of gating result)
    cvt_bf16_kernel<<<((size_t)Tn * Dd / 2 + 255) / 256, 256, 0, stream>>>(x, xbf, Tn * Dd / 2);
    transpose_cvt_kernel<<<dim3(Hh / 32, Dd / 32, Ee), dim3(32, 8), 0, stream>>>(w1, w1t, Dd, Hh);
    transpose_cvt_kernel<<<dim3(Dd / 32, Hh / 32, Ee), dim3(32, 8), 0, stream>>>(w2, w2t, Hh, Dd);

    // stage 1: h = gelu(x @ w1 + b1) per routed (token, expert)
    moe_gemm_kernel<Dd, 1><<<dim3(Hh / 256, Tn / 128, Ee), 256, 0, stream>>>(
        xbf, w1t, b1, list, cnt, hbuf, (float*)nullptr, Hh);
    // stage 2: y = h @ w2 + b2
    moe_gemm_kernel<Hh, 2><<<dim3(Dd / 256, Tn / 128, Ee), 256, 0, stream>>>(
        hbuf, w2t, b2, list, cnt, (unsigned short*)nullptr, ybuf, Dd);

    combine_kernel<<<((size_t)Tn * Dd / 4 + 255) / 256, 256, 0, stream>>>(ybuf, topk_w, out, Tn * Dd);
}
